// EvoformerIteration_36172214566944
// MI455X (gfx1250) — matrix operand order repaired and checked by reading, not compiled, not run on hardware
//
#include <hip/hip_runtime.h>
#include <hip/hip_bf16.h>
#include <math.h>

// ---------------------------------------------------------------------------
// Evoformer iteration for MI455X (gfx1250), wave32 + WMMA f32_16x16x32_f16.
// All GEMMs / attention matmuls / triangle einsums go through v_wmma.
// ---------------------------------------------------------------------------

typedef __attribute__((ext_vector_type(16))) _Float16 v16h;
typedef __attribute__((ext_vector_type(8)))  float    v8f;

__device__ __forceinline__ v8f wmma16(v16h a, v16h b, v8f c) {
  return __builtin_amdgcn_wmma_f32_16x16x32_f16(false, a, false, b, (short)0, c,
                                                false, false);
}

__device__ __forceinline__ float fast_rcp(float x) { return __builtin_amdgcn_rcpf(x); }
__device__ __forceinline__ float fast_sigmoid(float x) {
  return fast_rcp(1.f + __expf(-x));
}

// A fragment (16x32, 16-bit): lane&15 = row M; element e -> K offset:
__device__ __forceinline__ int a_kk(int e, int half) {
  return (e < 8) ? (half * 8 + e) : (16 + half * 8 + (e - 8));
}
// B fragment (32x16, 16-bit): lane&15 = col N; element e -> same K offset as A.
__device__ __forceinline__ int b_kk(int e, int half) { return a_kk(e, half); }

__device__ __forceinline__ v16h cvt16(const float* f) {
  v16h r;
#pragma unroll
  for (int i = 0; i < 16; ++i) r[i] = (_Float16)f[i];
  return r;
}

// A frag via 4x b128: two contiguous 8-float runs at k0+half*8 and k0+16+half*8
__device__ __forceinline__ v16h load_a_frag(const float* row, int k0, int half) {
  float fa[16];
  const float4* p0 = reinterpret_cast<const float4*>(row + k0 + half * 8);
  const float4* p1 = reinterpret_cast<const float4*>(row + k0 + 16 + half * 8);
  reinterpret_cast<float4*>(fa)[0] = p0[0];
  reinterpret_cast<float4*>(fa)[1] = p0[1];
  reinterpret_cast<float4*>(fa)[2] = p1[0];
  reinterpret_cast<float4*>(fa)[3] = p1[1];
  return cvt16(fa);
}
// B frag via 4x b128: two contiguous 8-float runs at k0+half*8 and k0+16+half*8
__device__ __forceinline__ v16h load_b_frag(const float* row, int k0, int half) {
  float fb[16];
  const float4* p0 = reinterpret_cast<const float4*>(row + k0 + half * 8);
  const float4* p1 = reinterpret_cast<const float4*>(row + k0 + 16 + half * 8);
  reinterpret_cast<float4*>(fb)[0] = p0[0];
  reinterpret_cast<float4*>(fb)[1] = p0[1];
  reinterpret_cast<float4*>(fb)[2] = p1[0];
  reinterpret_cast<float4*>(fb)[3] = p1[1];
  return cvt16(fb);
}

// ---------------------------------------------------------------------------
// Generic GEMM: Y[M,N] = act(X[M,K] @ W[N,K]^T + bias) * mulv + res
// 128-thread blocks = 4 waves, each wave owns one 16x16 tile (4 M-tiles/block).
// K multiple of 32, N multiple of 16, M multiple of 64.
// ---------------------------------------------------------------------------
template <int ACT, bool HB, bool HR, bool HMUL>
__global__ void gemm_wmma(const float* __restrict__ X, const float* __restrict__ W,
                          const float* __restrict__ bias, const float* __restrict__ res,
                          const float* __restrict__ mulv, float* __restrict__ Y,
                          int M, int N, int K) {
  const int lane = threadIdx.x & 31;
  const int wv = threadIdx.x >> 5;
  const int half = lane >> 4, q = lane & 15;
  const int n0 = blockIdx.x * 16;
  const int m0 = (blockIdx.y * 4 + wv) * 16;
  v8f acc = {0.f, 0.f, 0.f, 0.f, 0.f, 0.f, 0.f, 0.f};
  const float* xr = X + (size_t)(m0 + q) * K;
  const float* wr = W + (size_t)(n0 + q) * K;
  for (int k0 = 0; k0 < K; k0 += 32) {
    v16h a = load_a_frag(xr, k0, half);
    v16h b = load_b_frag(wr, k0, half);
    acc = wmma16(a, b, acc);
  }
  const int n = n0 + q;
  float bv = 0.f;
  if constexpr (HB) bv = bias[n];
#pragma unroll
  for (int j = 0; j < 8; ++j) {
    const int m = m0 + j + 8 * half;
    float v = acc[j];
    if constexpr (HB) v += bv;  // no spurious +0 in the no-bias variants
    if constexpr (ACT == 1) v = v > 0.f ? v : 0.f;
    if constexpr (ACT == 2) v = fast_sigmoid(v);
    const size_t idx = (size_t)m * N + n;
    if constexpr (HMUL) v *= mulv[idx];
    if constexpr (HR) v += res[idx];
    Y[idx] = v;
  }
}

// ---------------------------------------------------------------------------
// Fused attention (flash-style, logits never hit HBM). One wave handles a
// (group g, head h, 16-query tile). ch == 32. Scores via WMMA, softmax in
// LDS, P@V via WMMA. bias1[g,k] (key mask), bias2[h,q,k] via strides.
// Templated on bias presence: straight-line score epilogue.
// ---------------------------------------------------------------------------
template <bool B1, bool B2>
__global__ void attn_fused(const float* __restrict__ Qp, const float* __restrict__ Kp,
                           const float* __restrict__ Vp, float* __restrict__ Op,
                           int Lq, int Lk, int rowstride, float scale,
                           const float* __restrict__ bias1, int b1pitch,
                           const float* __restrict__ bias2, int b2h, int b2q, int b2k) {
  extern __shared__ float sc[];  // [16][Lk]
  const int lane = threadIdx.x, half = lane >> 4, q = lane & 15;
  const int q0 = blockIdx.x * 16, h = blockIdx.y, g = blockIdx.z;
  const int ch = 32;

  v16h aQ;
  {
    const float* qr = Qp + (size_t)(g * Lq + q0 + q) * rowstride + h * ch;
    float fa[16];
    const float4* p0 = reinterpret_cast<const float4*>(qr + half * 8);
    const float4* p1 = reinterpret_cast<const float4*>(qr + 16 + half * 8);
    reinterpret_cast<float4*>(fa)[0] = p0[0];
    reinterpret_cast<float4*>(fa)[1] = p0[1];
    reinterpret_cast<float4*>(fa)[2] = p1[0];
    reinterpret_cast<float4*>(fa)[3] = p1[1];
#pragma unroll
    for (int e = 0; e < 16; ++e) aQ[e] = (_Float16)(fa[e] * scale);
  }
  // ---- scores: S[16 x Lk] ----
  for (int k0 = 0; k0 < Lk; k0 += 16) {
    const int kk = k0 + q;  // this lane's key column
    const float* kr = Kp + (size_t)(g * Lk + kk) * rowstride + h * ch;
    v16h bK = load_b_frag(kr, 0, half);
    v8f z = {0.f, 0.f, 0.f, 0.f, 0.f, 0.f, 0.f, 0.f};
    v8f s = wmma16(aQ, bK, z);
    float b1v = 0.f;
    if constexpr (B1) b1v = bias1[(size_t)g * b1pitch + kk];  // j-invariant
    const float* b2p = nullptr;
    if constexpr (B2) b2p = bias2 + (size_t)h * b2h + (size_t)kk * b2k + (size_t)q0 * b2q;
#pragma unroll
    for (int j = 0; j < 8; ++j) {
      const int m = j + 8 * half;
      float v = s[j];
      if constexpr (B1) v += b1v;
      if constexpr (B2) v += b2p[(size_t)m * b2q];
      sc[m * Lk + kk] = v;
    }
  }
  __syncthreads();
  // ---- softmax, 2 lanes per row ----
  {
    const int row = lane & 15, part = lane >> 4;
    const int kb = part * (Lk / 2), ke = kb + Lk / 2;
    float mx = -1e30f;
    for (int k = kb; k < ke; ++k) mx = fmaxf(mx, sc[row * Lk + k]);
    mx = fmaxf(mx, __shfl_xor(mx, 16));
    float sum = 0.f;
    for (int k = kb; k < ke; ++k) {
      float e = __expf(sc[row * Lk + k] - mx);
      sc[row * Lk + k] = e;
      sum += e;
    }
    sum += __shfl_xor(sum, 16);
    const float inv = fast_rcp(sum);
    for (int k = kb; k < ke; ++k) sc[row * Lk + k] *= inv;
  }
  __syncthreads();
  // ---- O = P @ V (two 16-wide column halves of ch) ----
  v8f o0 = {0.f, 0.f, 0.f, 0.f, 0.f, 0.f, 0.f, 0.f};
  v8f o1 = {0.f, 0.f, 0.f, 0.f, 0.f, 0.f, 0.f, 0.f};
  for (int k0 = 0; k0 < Lk; k0 += 32) {
    v16h aP = load_a_frag(sc + q * Lk, k0, half);  // ds_load_b128 path
    v16h b0, b1f;
#pragma unroll
    for (int e = 0; e < 16; ++e) {
      const float* vr = Vp + (size_t)(g * Lk + k0 + b_kk(e, half)) * rowstride + h * ch;
      b0[e] = (_Float16)vr[q];
      b1f[e] = (_Float16)vr[16 + q];
    }
    o0 = wmma16(aP, b0, o0);
    o1 = wmma16(aP, b1f, o1);
  }
#pragma unroll
  for (int j = 0; j < 8; ++j) {
    const int m = q0 + j + 8 * half;
    float* orow = Op + (size_t)(g * Lq + m) * rowstride + h * ch;
    orow[q] = o0[j];
    orow[16 + q] = o1[j];
  }
}

// ---------------------------------------------------------------------------
// Triangle multiplication einsum: per channel c, O_c[i,j] = sum_k A_c * B_c.
// AT/BT (compile-time) select (row,k) vs (k,row) indexing into [R,R,C].
//   tmo ('bikc,bjkc'): tri_gemm<0,0>(lp, rp)
//   tmi ('bkjc,bkic'): tri_gemm<1,1>(rp, lp)
// 4 waves per block, one channel per wave.
// ---------------------------------------------------------------------------
template <int AT, int BT>
__global__ void tri_gemm(const float* __restrict__ Ap, const float* __restrict__ Bp,
                         float* __restrict__ Oc, int Rdim, int C) {
  const int lane = threadIdx.x & 31, half = lane >> 4, q = lane & 15;
  const int j0 = blockIdx.x * 16, i0 = blockIdx.y * 16;
  const int c = blockIdx.z * 4 + (threadIdx.x >> 5);
  v8f acc = {0.f, 0.f, 0.f, 0.f, 0.f, 0.f, 0.f, 0.f};
  for (int k0 = 0; k0 < Rdim; k0 += 32) {
    v16h a, b;
#pragma unroll
    for (int e = 0; e < 16; ++e) {
      const int ka = k0 + a_kk(e, half);
      const int kb = k0 + b_kk(e, half);
      size_t ai, bi;
      if constexpr (AT) ai = (size_t)ka * Rdim + (i0 + q);
      else              ai = (size_t)(i0 + q) * Rdim + ka;
      if constexpr (BT) bi = (size_t)kb * Rdim + (j0 + q);
      else              bi = (size_t)(j0 + q) * Rdim + kb;
      a[e] = (_Float16)Ap[ai * C + c];
      b[e] = (_Float16)Bp[bi * C + c];
    }
    acc = wmma16(a, b, acc);
  }
#pragma unroll
  for (int j = 0; j < 8; ++j) {
    const int i = i0 + j + 8 * half, jj = j0 + q;
    Oc[((size_t)i * Rdim + jj) * C + c] = acc[j];
  }
}

// ---------------------------------------------------------------------------
// LayerNorm: one wave per row, C in {128, 256}. In-place safe.
// ---------------------------------------------------------------------------
__global__ void layernorm_k(const float* __restrict__ X, const float* __restrict__ g,
                            const float* __restrict__ b, float* __restrict__ Y, int C) {
  const int row = blockIdx.x, lane = threadIdx.x;
  const float* x = X + (size_t)row * C;
  const int nv = C / 32;
  float vals[8];
  float s = 0.f;
  for (int i = 0; i < nv; ++i) { vals[i] = x[lane + i * 32]; s += vals[i]; }
#pragma unroll
  for (int o = 16; o > 0; o >>= 1) s += __shfl_xor(s, o);
  const float mean = s / C;
  float vs = 0.f;
  for (int i = 0; i < nv; ++i) { float d = vals[i] - mean; vs += d * d; }
#pragma unroll
  for (int o = 16; o > 0; o >>= 1) vs += __shfl_xor(vs, o);
  const float inv = rsqrtf(vs / C + 1e-5f);
  float* y = Y + (size_t)row * C;
  for (int i = 0; i < nv; ++i) {
    const int c = lane + i * 32;
    y[c] = (vals[i] - mean) * inv * g[c] + b[c];
  }
}

// Small-N linear (pair-bias projections, N = 4 or 8).
__global__ void linear_small(const float* __restrict__ X, const float* __restrict__ W,
                             float* __restrict__ Y, int M, int N, int K) {
  const int idx = blockIdx.x * blockDim.x + threadIdx.x;
  if (idx >= M * N) return;
  const int m = idx / N, n = idx % N;
  const float* x = X + (size_t)m * K;
  const float* w = W + (size_t)n * K;
  float acc = 0.f;
  for (int k = 0; k < K; ++k) acc += x[k] * w[k];
  Y[idx] = acc;
}

__global__ void ew_mul(float* __restrict__ A, const float* __restrict__ B, size_t n) {
  const size_t i = (size_t)blockIdx.x * blockDim.x + threadIdx.x;
  if (i < n) A[i] *= B[i];
}
__global__ void row_scale(float* __restrict__ A, const float* __restrict__ s, int M, int N) {
  const size_t i = (size_t)blockIdx.x * blockDim.x + threadIdx.x;
  if (i < (size_t)M * N) A[i] *= s[i / N];
}
__global__ void mask_bias(const float* __restrict__ mask, float* __restrict__ out,
                          int Rows, int L, int s0, int s1) {
  const int idx = blockIdx.x * blockDim.x + threadIdx.x;
  if (idx >= Rows * L) return;
  const int i = idx / L, k = idx % L;
  out[idx] = 1e9f * (mask[(size_t)i * s0 + (size_t)k * s1] - 1.f);
}
// out[(j*A + i)*C + c] = in[(i*Bd + j)*C + c]
__global__ void transpose_c(const float* __restrict__ in, float* __restrict__ out,
                            int A, int Bd, int C) {
  const size_t idx = (size_t)blockIdx.x * blockDim.x + threadIdx.x;
  if (idx >= (size_t)A * Bd * C) return;
  const int c = idx % C;
  const size_t t = idx / C;
  const int j = t % Bd, i = t / Bd;
  out[((size_t)j * A + i) * C + c] = in[idx];
}
// dst[(i*Bd + j)*C + c] += src[(j*A + i)*C + c]
__global__ void add_t(float* __restrict__ dst, const float* __restrict__ src,
                      int A, int Bd, int C) {
  const size_t idx = (size_t)blockIdx.x * blockDim.x + threadIdx.x;
  if (idx >= (size_t)A * Bd * C) return;
  const int c = idx % C;
  const size_t t = idx / C;
  const int j = t % Bd, i = t / Bd;
  dst[idx] += src[((size_t)j * A + i) * C + c];
}
__global__ void opm_norm_k(const float* __restrict__ mask, float* __restrict__ norm,
                           int S_, int R_) {
  const int idx = blockIdx.x * blockDim.x + threadIdx.x;
  if (idx >= R_ * R_) return;
  const int i = idx / R_, j = idx % R_;
  float s = 0.f;
  for (int m = 0; m < S_; ++m) s += mask[m * R_ + i] * mask[m * R_ + j];
  norm[idx] = s;
}

// ---------------------------------------------------------------------------
// Outer-product-mean, fused per (i,j): E[32,32] = sum_m la[m,i,:]⊗ra[m,j,:],
// then project 1024 -> 128 and pair[i,j] += proj/(norm+1e-3). 128 threads.
// ---------------------------------------------------------------------------
__global__ void opm_fuse(const float* __restrict__ la, const float* __restrict__ ra,
                         const float* __restrict__ Wp, const float* __restrict__ bp,
                         const float* __restrict__ norm, float* __restrict__ pair,
                         int S_, int R_) {
  __shared__ float LA[64 * 32];
  __shared__ float RA[64 * 32];
  __shared__ float E[1024];
  const int ij = blockIdx.x;
  const int i = ij / R_, j = ij % R_;
  const int t = threadIdx.x;
  for (int u = t; u < S_ * 32; u += 128) {
    const int m = u >> 5, x = u & 31;
    LA[u] = la[((size_t)m * R_ + i) * 32 + x];
    RA[u] = ra[((size_t)m * R_ + j) * 32 + x];
  }
  __syncthreads();
  for (int u = t; u < 1024; u += 128) {
    const int x = u >> 5, y = u & 31;
    float s = 0.f;
    for (int m = 0; m < S_; ++m) s += LA[m * 32 + x] * RA[m * 32 + y];
    E[u] = s;
  }
  __syncthreads();
  float acc = bp[t];
  const float* w = Wp + (size_t)t * 1024;
  for (int u = 0; u < 1024; ++u) acc += w[u] * E[u];
  pair[(size_t)ij * 128 + t] += acc / (norm[ij] + 1e-3f);
}

// ---------------------------------------------------------------------------
// Host orchestration
// ---------------------------------------------------------------------------
extern "C" void kernel_launch(void* const* d_in, const int* in_sizes, int n_in,
                              void* d_out, int out_size, void* d_ws, size_t ws_size,
                              hipStream_t stream) {
  (void)in_sizes; (void)n_in; (void)out_size; (void)ws_size;
  constexpr int S_ = 64, R_ = 256, CM = 256, CZ = 128, HM = 8, HZ = 4, CH = 32;
  constexpr int SR = S_ * R_;   // 16384
  constexpr int RR = R_ * R_;   // 65536
  constexpr size_t MSA_E = (size_t)SR * CM;    // 4,194,304
  constexpr size_t PAIR_E = (size_t)RR * CZ;   // 8,388,608
  const float sc_att = 1.f / sqrtf((float)CH);

  const float* in_msa = (const float*)d_in[0];
  const float* in_pair = (const float*)d_in[1];
  const float* msa_mask = (const float*)d_in[2];
  const float* pair_mask = (const float*)d_in[3];

  int pi = 4;
  auto P = [&]() -> const float* { return (const float*)d_in[pi++]; };
  // params in _make_params() insertion order
  const float *row_qn_g = P(), *row_qn_b = P();
  const float *row_2dn_g = P(), *row_2dn_b = P();
  const float *row_2dw = P();
  const float *row_qw = P(), *row_kw = P(), *row_vw = P();
  const float *row_gw = P(), *row_gb = P(), *row_ow = P(), *row_ob = P();
  const float *col_qn_g = P(), *col_qn_b = P();
  const float *col_qw = P(), *col_kw = P(), *col_vw = P();
  const float *col_gw = P(), *col_gb = P(), *col_ow = P(), *col_ob = P();
  const float *mt_ln_g = P(), *mt_ln_b = P();
  const float *mt_w1 = P(), *mt_b1 = P(), *mt_w2 = P(), *mt_b2 = P();
  const float *opm_ln_g = P(), *opm_ln_b = P();
  const float *opm_lw = P(), *opm_lb = P(), *opm_rw = P(), *opm_rb = P();
  const float *opm_ow = P(), *opm_ob = P();
  struct TriP { const float *ln_g, *ln_b, *lp_w, *lp_b, *rp_w, *rp_b,
                            *lg_w, *lg_b, *rg_w, *rg_b, *cln_g, *cln_b,
                            *op_w, *op_b, *g_w, *g_b; };
  TriP tp[2];
  for (int t = 0; t < 2; ++t) {
    tp[t].ln_g = P(); tp[t].ln_b = P();
    tp[t].lp_w = P(); tp[t].lp_b = P(); tp[t].rp_w = P(); tp[t].rp_b = P();
    tp[t].lg_w = P(); tp[t].lg_b = P(); tp[t].rg_w = P(); tp[t].rg_b = P();
    tp[t].cln_g = P(); tp[t].cln_b = P(); tp[t].op_w = P(); tp[t].op_b = P();
    tp[t].g_w = P(); tp[t].g_b = P();
  }
  struct TAP { const float *qn_g, *qn_b, *dw, *qw, *kw, *vw, *gw, *gb, *ow, *ob; };
  TAP ta[2];
  for (int t = 0; t < 2; ++t) {
    ta[t].qn_g = P(); ta[t].qn_b = P(); ta[t].dw = P();
    ta[t].qw = P(); ta[t].kw = P(); ta[t].vw = P();
    ta[t].gw = P(); ta[t].gb = P(); ta[t].ow = P(); ta[t].ob = P();
  }
  const float *pt_ln_g = P(), *pt_ln_b = P();
  const float *pt_w1 = P(), *pt_b1 = P(), *pt_w2 = P(), *pt_b2 = P();

  // Live outputs
  float* msa = (float*)d_out;
  float* pr = msa + MSA_E;

  // Workspace carve (~338 MB of floats)
  float* LN = (float*)d_ws;            // 8.39M
  float* Qb = LN + PAIR_E;
  float* Kb = Qb + PAIR_E;
  float* Vb = Kb + PAIR_E;
  float* GT = Vb + PAIR_E;
  float* Ob = GT + PAIR_E;
  float* Hb = Ob + PAIR_E;             // 33.55M floats; also hosts Tb/T2b
  float* B2 = Hb + (size_t)33554432;
  float* BIAS1 = B2 + 524288;
  float* NORMb = BIAS1 + 65536;
  float* Tb = Hb;
  float* T2b = Hb + PAIR_E;

  // GEMM dispatch over the combinations actually used.
  auto gemm = [&](const float* X, const float* W, const float* bias, const float* res,
                  const float* mulv, float* Y, int M, int N, int K, int act) {
    dim3 grd(N / 16, M / 64);
    dim3 blk(128);
    if (!bias && !res && !mulv && act == 0)
      gemm_wmma<0, false, false, false><<<grd, blk, 0, stream>>>(X, W, bias, res, mulv, Y, M, N, K);
    else if (bias && !res && !mulv && act == 2)
      gemm_wmma<2, true, false, false><<<grd, blk, 0, stream>>>(X, W, bias, res, mulv, Y, M, N, K);
    else if (bias && res && mulv && act == 0)
      gemm_wmma<0, true, true, true><<<grd, blk, 0, stream>>>(X, W, bias, res, mulv, Y, M, N, K);
    else if (bias && res && !mulv && act == 0)
      gemm_wmma<0, true, true, false><<<grd, blk, 0, stream>>>(X, W, bias, res, mulv, Y, M, N, K);
    else if (bias && !res && !mulv && act == 1)
      gemm_wmma<1, true, false, false><<<grd, blk, 0, stream>>>(X, W, bias, res, mulv, Y, M, N, K);
    else
      gemm_wmma<0, true, false, false><<<grd, blk, 0, stream>>>(X, W, bias, res, mulv, Y, M, N, K);
  };
  auto lnorm = [&](const float* X, const float* g, const float* b, float* Y, int rows, int C) {
    layernorm_k<<<rows, 32, 0, stream>>>(X, g, b, Y, C);
  };
  auto ew = [&](float* A, const float* B, size_t n) {
    ew_mul<<<(unsigned)((n + 255) / 256), 256, 0, stream>>>(A, B, n);
  };

  hipMemcpyAsync(msa, in_msa, MSA_E * sizeof(float), hipMemcpyDeviceToDevice, stream);
  hipMemcpyAsync(pr, in_pair, PAIR_E * sizeof(float), hipMemcpyDeviceToDevice, stream);

  // ===== 1. MSA row attention with pair bias =====
  lnorm(pr, row_2dn_g, row_2dn_b, LN, RR, CZ);
  linear_small<<<(RR * HM + 255) / 256, 256, 0, stream>>>(LN, row_2dw, B2, RR, HM, CZ);
  lnorm(msa, row_qn_g, row_qn_b, LN, SR, CM);
  gemm(LN, row_qw, nullptr, nullptr, nullptr, Qb, SR, CM, CM, 0);
  gemm(LN, row_kw, nullptr, nullptr, nullptr, Kb, SR, CM, CM, 0);
  gemm(LN, row_vw, nullptr, nullptr, nullptr, Vb, SR, CM, CM, 0);
  gemm(LN, row_gw, row_gb, nullptr, nullptr, GT, SR, CM, CM, 2);
  mask_bias<<<(S_ * R_ + 255) / 256, 256, 0, stream>>>(msa_mask, BIAS1, S_, R_, R_, 1);
  attn_fused<true, true><<<dim3(R_ / 16, HM, S_), 32, 16 * R_ * sizeof(float), stream>>>(
      Qb, Kb, Vb, Ob, R_, R_, HM * CH, sc_att, BIAS1, R_, B2, 1, R_ * HM, HM);
  ew(Ob, GT, MSA_E);
  gemm(Ob, row_ow, row_ob, msa, nullptr, msa, SR, CM, CM, 0);

  // ===== 2. MSA column attention (transposed) =====
  transpose_c<<<(unsigned)((MSA_E + 255) / 256), 256, 0, stream>>>(msa, Tb, S_, R_, CM);
  lnorm(Tb, col_qn_g, col_qn_b, LN, SR, CM);
  gemm(LN, col_qw, nullptr, nullptr, nullptr, Qb, SR, CM, CM, 0);
  gemm(LN, col_kw, nullptr, nullptr, nullptr, Kb, SR, CM, CM, 0);
  gemm(LN, col_vw, nullptr, nullptr, nullptr, Vb, SR, CM, CM, 0);
  gemm(LN, col_gw, col_gb, nullptr, nullptr, GT, SR, CM, CM, 2);
  mask_bias<<<(R_ * S_ + 255) / 256, 256, 0, stream>>>(msa_mask, BIAS1, R_, S_, 1, R_);
  attn_fused<true, false><<<dim3(S_ / 16, HM, R_), 32, 16 * S_ * sizeof(float), stream>>>(
      Qb, Kb, Vb, Ob, S_, S_, HM * CH, sc_att, BIAS1, S_, nullptr, 0, 0, 0);
  ew(Ob, GT, MSA_E);
  gemm(Ob, col_ow, col_ob, nullptr, nullptr, T2b, SR, CM, CM, 0);
  add_t<<<(unsigned)((MSA_E + 255) / 256), 256, 0, stream>>>(msa, T2b, S_, R_, CM);

  // ===== 3. MSA transition =====
  lnorm(msa, mt_ln_g, mt_ln_b, LN, SR, CM);
  gemm(LN, mt_w1, mt_b1, nullptr, nullptr, Hb, SR, 4 * CM, CM, 1);
  gemm(Hb, mt_w2, mt_b2, msa, nullptr, msa, SR, CM, 4 * CM, 0);

  // ===== 4. Outer product mean =====
  lnorm(msa, opm_ln_g, opm_ln_b, LN, SR, CM);
  gemm(LN, opm_lw, opm_lb, nullptr, nullptr, Qb, SR, 32, CM, 0);
  row_scale<<<(SR * 32 + 255) / 256, 256, 0, stream>>>(Qb, msa_mask, SR, 32);
  gemm(LN, opm_rw, opm_rb, nullptr, nullptr, Kb, SR, 32, CM, 0);
  row_scale<<<(SR * 32 + 255) / 256, 256, 0, stream>>>(Kb, msa_mask, SR, 32);
  opm_norm_k<<<(RR + 255) / 256, 256, 0, stream>>>(msa_mask, NORMb, S_, R_);
  opm_fuse<<<RR, 128, 0, stream>>>(Qb, Kb, opm_ow, opm_ob, NORMb, pr, S_, R_);

  // ===== 5/6. Triangle multiplication outgoing ('tmo') / ingoing ('tmi') =====
  for (int t = 0; t < 2; ++t) {
    const TriP& T = tp[t];
    lnorm(pr, T.ln_g, T.ln_b, LN, RR, CZ);
    gemm(LN, T.lp_w, T.lp_b, nullptr, nullptr, Qb, RR, CZ, CZ, 0);
    gemm(LN, T.lg_w, T.lg_b, nullptr, nullptr, GT, RR, CZ, CZ, 2);
    ew(Qb, GT, PAIR_E);
    row_scale<<<(unsigned)((PAIR_E + 255) / 256), 256, 0, stream>>>(Qb, pair_mask, RR, CZ);
    gemm(LN, T.rp_w, T.rp_b, nullptr, nullptr, Kb, RR, CZ, CZ, 0);
    gemm(LN, T.rg_w, T.rg_b, nullptr, nullptr, GT, RR, CZ, CZ, 2);
    ew(Kb, GT, PAIR_E);
    row_scale<<<(unsigned)((PAIR_E + 255) / 256), 256, 0, stream>>>(Kb, pair_mask, RR, CZ);
    gemm(LN, T.g_w, T.g_b, nullptr, nullptr, GT, RR, CZ, CZ, 2);  // output gate
    if (t == 0)  // 'bikc,bjkc->bijc'
      tri_gemm<0, 0><<<dim3(R_ / 16, R_ / 16, CZ / 4), 128, 0, stream>>>(Qb, Kb, Ob, R_, CZ);
    else         // 'bkjc,bkic->bijc'
      tri_gemm<1, 1><<<dim3(R_ / 16, R_ / 16, CZ / 4), 128, 0, stream>>>(Kb, Qb, Ob, R_, CZ);
    lnorm(Ob, T.cln_g, T.cln_b, Ob, RR, CZ);
    gemm(Ob, T.op_w, T.op_b, pr, GT, pr, RR, CZ, CZ, 0);
  }

  // ===== 7. Triangle attention, starting node ('tas') =====
  {
    const TAP& A = ta[0];
    lnorm(pr, A.qn_g, A.qn_b, LN, RR, CZ);
    linear_small<<<(RR * HZ + 255) / 256, 256, 0, stream>>>(LN, A.dw, B2, RR, HZ, CZ);
    gemm(LN, A.qw, nullptr, nullptr, nullptr, Qb, RR, CZ, CZ, 0);
    gemm(LN, A.kw, nullptr, nullptr, nullptr, Kb, RR, CZ, CZ, 0);
    gemm(LN, A.vw, nullptr, nullptr, nullptr, Vb, RR, CZ, CZ, 0);
    gemm(LN, A.gw, A.gb, nullptr, nullptr, GT, RR, CZ, CZ, 2);
    mask_bias<<<(RR + 255) / 256, 256, 0, stream>>>(pair_mask, BIAS1, R_, R_, R_, 1);
    attn_fused<true, true><<<dim3(R_ / 16, HZ, R_), 32, 16 * R_ * sizeof(float), stream>>>(
        Qb, Kb, Vb, Ob, R_, R_, HZ * CH, sc_att, BIAS1, R_, B2, 1, R_ * HZ, HZ);
    ew(Ob, GT, PAIR_E);
    gemm(Ob, A.ow, A.ob, pr, nullptr, pr, RR, CZ, CZ, 0);
  }

  // ===== 8. Triangle attention, ending node ('tae', transposed) =====
  {
    const TAP& A = ta[1];
    transpose_c<<<(unsigned)((PAIR_E + 255) / 256), 256, 0, stream>>>(pr, Tb, R_, R_, CZ);
    lnorm(Tb, A.qn_g, A.qn_b, LN, RR, CZ);
    linear_small<<<(RR * HZ + 255) / 256, 256, 0, stream>>>(LN, A.dw, B2, RR, HZ, CZ);
    gemm(LN, A.qw, nullptr, nullptr, nullptr, Qb, RR, CZ, CZ, 0);
    gemm(LN, A.kw, nullptr, nullptr, nullptr, Kb, RR, CZ, CZ, 0);
    gemm(LN, A.vw, nullptr, nullptr, nullptr, Vb, RR, CZ, CZ, 0);
    gemm(LN, A.gw, A.gb, nullptr, nullptr, GT, RR, CZ, CZ, 2);
    mask_bias<<<(RR + 255) / 256, 256, 0, stream>>>(pair_mask, BIAS1, R_, R_, 1, R_);
    attn_fused<true, true><<<dim3(R_ / 16, HZ, R_), 32, 16 * R_ * sizeof(float), stream>>>(
        Qb, Kb, Vb, Ob, R_, R_, HZ * CH, sc_att, BIAS1, R_, B2, 1, R_ * HZ, HZ);
    ew(Ob, GT, PAIR_E);
    gemm(Ob, A.ow, A.ob, nullptr, nullptr, T2b, RR, CZ, CZ, 0);
    add_t<<<(unsigned)((PAIR_E + 255) / 256), 256, 0, stream>>>(pr, T2b, R_, R_, CZ);
  }

  // ===== 9. Pair transition =====
  lnorm(pr, pt_ln_g, pt_ln_b, LN, RR, CZ);
  gemm(LN, pt_w1, pt_b1, nullptr, nullptr, Hb, RR, 4 * CZ, CZ, 1);
  gemm(Hb, pt_w2, pt_b2, pr, nullptr, pr, RR, CZ, 4 * CZ, 0);
}
